// LSTM_42640435314705
// MI455X (gfx1250) — compile-verified, matrix operand order checked by reading, not run on hardware
//
#include <hip/hip_runtime.h>
#include <hip/hip_bf16.h>

#define BB 128
#define SS 512
#define DD 128
#define HH 1024
#define NC 10
#define TT (SS - 1)

typedef __attribute__((ext_vector_type(16))) __bf16 v16bf;
typedef __attribute__((ext_vector_type(8)))  __bf16 v8bf;
typedef __attribute__((ext_vector_type(8)))  float  v8f;

__device__ __forceinline__ unsigned short f2bf(float f) {
    unsigned int u = __float_as_uint(f);
    u += 0x7FFFu + ((u >> 16) & 1u);   // round-to-nearest-even
    return (unsigned short)(u >> 16);
}

// Load a 16x32 bf16 WMMA fragment (A or B operand) per ISA 7.12.2 layout:
// lane<16 holds K {0..7, 16..23} of row (lane&15); lane>=16 holds K {8..15, 24..31}.
__device__ __forceinline__ v16bf load_frag(const unsigned short* __restrict__ rowptr, int k0) {
    v8bf lo = *(const v8bf*)(rowptr + k0);
    v8bf hi = *(const v8bf*)(rowptr + k0 + 16);
    return __builtin_shufflevector(lo, hi, 0,1,2,3,4,5,6,7,8,9,10,11,12,13,14,15);
}

struct FragSet { v16bf a, b0, b1, b2, b3; };

// ---------------- setup kernels ----------------

__global__ void k_conv_x(const float* __restrict__ x, unsigned short* __restrict__ xb, int n) {
    for (int i = blockIdx.x * blockDim.x + threadIdx.x; i < n; i += gridDim.x * blockDim.x)
        xb[i] = f2bf(x[i]);
}

__global__ void k_pack_wh(const float* __restrict__ w0, const float* __restrict__ w1,
                          const float* __restrict__ w2, const float* __restrict__ w3,
                          unsigned short* __restrict__ dst) {
    const int n = 4 * HH * HH;
    for (int i = blockIdx.x * blockDim.x + threadIdx.x; i < n; i += gridDim.x * blockDim.x) {
        int g = i / (HH * HH), r = i % (HH * HH);
        const float* w = (g == 0) ? w0 : (g == 1) ? w1 : (g == 2) ? w2 : w3;
        dst[i] = f2bf(w[r]);
    }
}

__global__ void k_pack_wx(const float* __restrict__ w0, const float* __restrict__ w1,
                          const float* __restrict__ w2, const float* __restrict__ w3,
                          unsigned short* __restrict__ dst) {
    const int n = 4 * HH * DD;
    for (int i = blockIdx.x * blockDim.x + threadIdx.x; i < n; i += gridDim.x * blockDim.x) {
        int g = i / (HH * DD), r = i % (HH * DD);
        const float* w = (g == 0) ? w0 : (g == 1) ? w1 : (g == 2) ? w2 : w3;
        dst[i] = f2bf(w[r]);
    }
}

__global__ void k_init_state(const float* __restrict__ h_init, const float* __restrict__ c_init,
                             unsigned short* __restrict__ hb0, float* __restrict__ cbuf,
                             float* __restrict__ hf) {
    const int n = BB * HH;
    for (int i = blockIdx.x * blockDim.x + threadIdx.x; i < n; i += gridDim.x * blockDim.x) {
        int h = i & (HH - 1);
        hb0[i] = f2bf(h_init[h]);
        cbuf[i] = c_init[h];
        hf[i] = h_init[h];
    }
}

// ---------------- fused LSTM step: 4-gate WMMA GEMM + cell update ----------------
// grid = 128 blocks (8 M-blocks x 16 N-blocks), 128 threads (4 waves, wave g = gate g).
// gates[b,n] = sum_k h_prev[b,k]*Wh[g][n,k] + sum_d x_t[b,d]*Wx[g][n,d]  (bias in pointwise)
// K-space unified: iterations 0..31 -> h_prev (K=1024), 32..35 -> x_t (K=128).

#define NIT (HH / 32 + DD / 32)   // 36

__global__ void __launch_bounds__(128)
k_lstm_step(const unsigned short* __restrict__ xb,
            const unsigned short* __restrict__ Whb,
            const unsigned short* __restrict__ Wxb,
            const float* __restrict__ bg, const float* __restrict__ bi,
            const float* __restrict__ bfv, const float* __restrict__ bo,
            const unsigned short* __restrict__ h_rd,
            unsigned short* __restrict__ h_wr,
            float* __restrict__ cbuf,
            float* __restrict__ hf,
            int t) {
    __shared__ float lds_g[4][16][64];

    const int lane = threadIdx.x & 31;
    const int g    = threadIdx.x >> 5;         // wave index == gate
    const int row  = lane & 15;
    const int klo  = (lane < 16) ? 0 : 8;
    const int m0   = (blockIdx.x & 7) * 16;    // batch tile
    const int n0   = (blockIdx.x >> 3) * 64;   // hidden tile

    // per-lane row base pointers (A: state/x rows; B: weight rows = output columns)
    const unsigned short* arow  = h_rd + (size_t)(m0 + row) * HH;
    const unsigned short* xrow  = xb + (size_t)(m0 + row) * (SS * DD) + (size_t)t * DD;
    const unsigned short* brow0 = Whb + ((size_t)g * HH + n0 +  0 + row) * HH;
    const unsigned short* brow1 = Whb + ((size_t)g * HH + n0 + 16 + row) * HH;
    const unsigned short* brow2 = Whb + ((size_t)g * HH + n0 + 32 + row) * HH;
    const unsigned short* brow3 = Whb + ((size_t)g * HH + n0 + 48 + row) * HH;
    const unsigned short* xw0   = Wxb + ((size_t)g * HH + n0 +  0 + row) * DD;
    const unsigned short* xw1   = Wxb + ((size_t)g * HH + n0 + 16 + row) * DD;
    const unsigned short* xw2   = Wxb + ((size_t)g * HH + n0 + 32 + row) * DD;
    const unsigned short* xw3   = Wxb + ((size_t)g * HH + n0 + 48 + row) * DD;

    auto load_set = [&](FragSet& s, int it) {
        if (it < HH / 32) {                    // recurrent part
            const int k0 = it * 32 + klo;
            s.a  = load_frag(arow,  k0);
            s.b0 = load_frag(brow0, k0);
            s.b1 = load_frag(brow1, k0);
            s.b2 = load_frag(brow2, k0);
            s.b3 = load_frag(brow3, k0);
        } else {                               // input part
            const int k0 = (it - HH / 32) * 32 + klo;
            s.a  = load_frag(xrow, k0);
            s.b0 = load_frag(xw0,  k0);
            s.b1 = load_frag(xw1,  k0);
            s.b2 = load_frag(xw2,  k0);
            s.b3 = load_frag(xw3,  k0);
        }
    };

    v8f acc0 = {}, acc1 = {}, acc2 = {}, acc3 = {};
    FragSet fA, fB;

    // software-pipelined ping-pong: next slice's 10 loads stream under current WMMAs
    load_set(fA, 0);
    for (int it = 0; it < NIT; it += 2) {
        load_set(fB, it + 1);                  // NIT even, it+1 always valid
        acc0 = __builtin_amdgcn_wmma_f32_16x16x32_bf16(false, fA.a, false, fA.b0, (short)0, acc0, false, false);
        acc1 = __builtin_amdgcn_wmma_f32_16x16x32_bf16(false, fA.a, false, fA.b1, (short)0, acc1, false, false);
        acc2 = __builtin_amdgcn_wmma_f32_16x16x32_bf16(false, fA.a, false, fA.b2, (short)0, acc2, false, false);
        acc3 = __builtin_amdgcn_wmma_f32_16x16x32_bf16(false, fA.a, false, fA.b3, (short)0, acc3, false, false);
        if (it + 2 < NIT) load_set(fA, it + 2);
        acc0 = __builtin_amdgcn_wmma_f32_16x16x32_bf16(false, fB.a, false, fB.b0, (short)0, acc0, false, false);
        acc1 = __builtin_amdgcn_wmma_f32_16x16x32_bf16(false, fB.a, false, fB.b1, (short)0, acc1, false, false);
        acc2 = __builtin_amdgcn_wmma_f32_16x16x32_bf16(false, fB.a, false, fB.b2, (short)0, acc2, false, false);
        acc3 = __builtin_amdgcn_wmma_f32_16x16x32_bf16(false, fB.a, false, fB.b3, (short)0, acc3, false, false);
    }

    // ---- spill gate tiles to LDS (C layout: VGPR r -> M = r + 8*(lane>=16), N = lane&15) ----
    const int mh = (lane >> 4) << 3;
    const int nc = lane & 15;
#pragma unroll
    for (int r = 0; r < 8; ++r) {
        lds_g[g][mh + r][ 0 + nc] = acc0[r];
        lds_g[g][mh + r][16 + nc] = acc1[r];
        lds_g[g][mh + r][32 + nc] = acc2[r];
        lds_g[g][mh + r][48 + nc] = acc3[r];
    }
    __syncthreads();

    // ---- pointwise LSTM cell update for this 16x64 tile ----
    for (int e = threadIdx.x; e < 16 * 64; e += 128) {
        const int m = e >> 6;
        const int n = e & 63;
        const int hcol = n0 + n;
        float G = lds_g[0][m][n] + bg[hcol];
        float I = lds_g[1][m][n] + bi[hcol];
        float F = lds_g[2][m][n] + bfv[hcol];
        float O = lds_g[3][m][n] + bo[hcol];
        float gg = tanhf(G);
        float ii = 1.0f / (1.0f + __expf(-I));
        float ff = 1.0f / (1.0f + __expf(-F));
        float oo = 1.0f / (1.0f + __expf(-O));
        const size_t idx = (size_t)(m0 + m) * HH + hcol;
        float c = cbuf[idx];
        c = gg * ii + c * ff;
        cbuf[idx] = c;
        float h = tanhf(c) * oo;
        hf[idx]  = h;
        h_wr[idx] = f2bf(h);
    }
}

// ---------------- final projection: p = h @ Wph.T + bp  (tiny: 128x10) ----------------
__global__ void k_proj(const float* __restrict__ hf, const float* __restrict__ Wph,
                       const float* __restrict__ bp, float* __restrict__ out) {
    int i = blockIdx.x * blockDim.x + threadIdx.x;
    if (i >= BB * NC) return;
    int b = i / NC, c = i % NC;
    const float* hp = hf  + (size_t)b * HH;
    const float* wp = Wph + (size_t)c * HH;
    float s = bp[c];
    for (int k = 0; k < HH; ++k) s += hp[k] * wp[k];
    out[i] = s;
}

extern "C" void kernel_launch(void* const* d_in, const int* in_sizes, int n_in,
                              void* d_out, int out_size, void* d_ws, size_t ws_size,
                              hipStream_t stream) {
    const float* x     = (const float*)d_in[0];
    const float* Wgh   = (const float*)d_in[1];
    const float* Wih   = (const float*)d_in[2];
    const float* Wfh   = (const float*)d_in[3];
    const float* Woh   = (const float*)d_in[4];
    const float* Wgx   = (const float*)d_in[5];
    const float* Wix   = (const float*)d_in[6];
    const float* Wfx   = (const float*)d_in[7];
    const float* Wox   = (const float*)d_in[8];
    const float* Wph   = (const float*)d_in[9];
    const float* bg    = (const float*)d_in[10];
    const float* bi    = (const float*)d_in[11];
    const float* bf    = (const float*)d_in[12];
    const float* bo    = (const float*)d_in[13];
    const float* bp    = (const float*)d_in[14];
    const float* h_init= (const float*)d_in[15];
    const float* c_init= (const float*)d_in[16];

    char* ws = (char*)d_ws;
    unsigned short* xb  = (unsigned short*)ws; ws += (size_t)BB * SS * DD * 2;   // 16.8 MB
    unsigned short* Whb = (unsigned short*)ws; ws += (size_t)4 * HH * HH * 2;    //  8.4 MB
    unsigned short* Wxb = (unsigned short*)ws; ws += (size_t)4 * HH * DD * 2;    //  1.0 MB
    unsigned short* hb0 = (unsigned short*)ws; ws += (size_t)BB * HH * 2;
    unsigned short* hb1 = (unsigned short*)ws; ws += (size_t)BB * HH * 2;
    float* cbuf = (float*)ws; ws += (size_t)BB * HH * 4;
    float* hf   = (float*)ws; ws += (size_t)BB * HH * 4;

    // one-time conversions (cheap, bandwidth-bound)
    k_conv_x   <<<1024, 256, 0, stream>>>(x, xb, BB * SS * DD);
    k_pack_wh  <<<1024, 256, 0, stream>>>(Wgh, Wih, Wfh, Woh, Whb);
    k_pack_wx  <<<256,  256, 0, stream>>>(Wgx, Wix, Wfx, Wox, Wxb);
    k_init_state<<<256, 256, 0, stream>>>(h_init, c_init, hb0, cbuf, hf);

    // recurrence: one fused WMMA step per timestep (kernel boundary = global sync on h)
    for (int t = 0; t < TT; ++t) {
        const unsigned short* hr = (t & 1) ? hb1 : hb0;
        unsigned short*       hw = (t & 1) ? hb0 : hb1;
        k_lstm_step<<<128, 128, 0, stream>>>(xb, Whb, Wxb, bg, bi, bf, bo,
                                             hr, hw, cbuf, hf, t);
    }

    k_proj<<<(BB * NC + 127) / 128, 128, 0, stream>>>(hf, Wph, bp, (float*)d_out);
}
